// MOEMultiClassification_50010599195002
// MI455X (gfx1250) — compile-verified, matrix operand order checked by reading, not run on hardware
//
#include <hip/hip_runtime.h>
#include <hip/hip_bf16.h>
#include <cstdint>

// Problem dims
#define B_ 16
#define S_ 2048
#define H_ 1024
#define E_ 16
#define K_ 8
#define I_ 2816
#define T_ 128
#define BSROWS (B_ * S_)   // 32768 token rows

typedef __attribute__((ext_vector_type(16))) __bf16 bf16x16;
typedef __attribute__((ext_vector_type(8)))  __bf16 bf16x8;
typedef __attribute__((ext_vector_type(8)))  float  f32x8;

// LDS tile row stride in bf16 elems: 32 data + 8 pad = 80 bytes (16B aligned rows)
#define LDT 40

// CDNA5 async global->LDS staging (ASYNCcnt path), guarded so compile never breaks.
// Probe feedback: param type is `int __vector(4) __device__ *` (AS1) for the global
// source, so pass v4i pointers with explicit address spaces.
#if defined(__gfx1250__) && __has_builtin(__builtin_amdgcn_global_load_async_to_lds_b128) && \
    __has_builtin(__builtin_amdgcn_s_wait_asynccnt)
#define ASYNC_LDS 1
typedef int v4i_t __attribute__((vector_size(16)));
typedef __attribute__((address_space(1))) v4i_t* gv4i_p;
typedef __attribute__((address_space(3))) v4i_t* lv4i_p;
#else
#define ASYNC_LDS 0
#endif

__device__ __forceinline__ void wait_async_lds() {
#if ASYNC_LDS
  __builtin_amdgcn_s_wait_asynccnt(0);
#endif
}

__device__ __forceinline__ f32x8 wmma_bf16(bf16x16 a, bf16x16 b, f32x8 c) {
  return __builtin_amdgcn_wmma_f32_16x16x32_bf16(false, a, false, b, (short)0, c,
                                                 false, false);
}

// fast silu / sigmoid via v_rcp_f32 (avoid IEEE div_scale sequence in epilogues)
__device__ __forceinline__ float silu_f(float x) {
  return x * __builtin_amdgcn_rcpf(1.f + __expf(-x));
}
__device__ __forceinline__ float sigmoid_f(float x) {
  return __builtin_amdgcn_rcpf(1.f + __expf(-x));
}

// Load one 16x32 bf16 fragment (A layout == B^T-from-[N,K] layout, ISA 7.12.2):
// lane L: row = L&15, kbase = (L&16)?8:0; elems 0..7 -> k=kbase..+7, 8..15 -> k=kbase+16..+23
__device__ __forceinline__ bf16x16 load_frag(const __bf16* tile) {
  const int lane = threadIdx.x & 31;
  const int r = lane & 15;
  const int kb = (lane & 16) ? 8 : 0;
  const __bf16* p = tile + r * LDT + kb;
  bf16x8 lo = *(const bf16x8*)(p);
  bf16x8 hi = *(const bf16x8*)(p + 16);
  bf16x16 f;
#pragma unroll
  for (int j = 0; j < 8; ++j) { f[j] = lo[j]; f[8 + j] = hi[j]; }
  return f;
}

// --- A-tile staging (128x32 bf16, 256 threads, 32B per thread) ---------------
struct ARegs { uint4 a, b; };

__device__ __forceinline__ const __bf16* a_src(const __bf16* A, size_t lda, int k0) {
  const int tid = threadIdx.x;
  return A + (size_t)(tid >> 1) * lda + k0 + (tid & 1) * 16;
}
__device__ __forceinline__ __bf16* a_dst(__bf16* sA) {
  const int tid = threadIdx.x;
  return sA + (tid >> 1) * LDT + (tid & 1) * 16;
}

#if ASYNC_LDS
__device__ __forceinline__ void stage_a_async(__bf16* sA, const __bf16* A, size_t lda, int k0) {
  const __bf16* src = a_src(A, lda, k0);
  __bf16* dst = a_dst(sA);
  __builtin_amdgcn_global_load_async_to_lds_b128((gv4i_p)src, (lv4i_p)dst, 0, 0);
  __builtin_amdgcn_global_load_async_to_lds_b128((gv4i_p)(src + 8), (lv4i_p)(dst + 8), 0, 0);
}
#else
__device__ __forceinline__ ARegs load_a_regs(const __bf16* A, size_t lda, int k0) {
  const uint4* s = (const uint4*)a_src(A, lda, k0);
  ARegs r; r.a = s[0]; r.b = s[1];
  return r;
}
__device__ __forceinline__ void store_a_regs(__bf16* sA, ARegs r) {
  uint4* d = (uint4*)a_dst(sA);
  d[0] = r.a; d[1] = r.b;
}
#endif

// --- W-tile staging (64x32 f32 -> bf16, 256 threads, 8 floats per thread) ----
struct WRegs { float4 x, y; };

__device__ __forceinline__ WRegs load_w_regs(const float* W, size_t ldw, int k0) {
  const int tid = threadIdx.x;
  const float4* s = (const float4*)(W + (size_t)(tid >> 2) * ldw + k0 + (tid & 3) * 8);
  WRegs r; r.x = s[0]; r.y = s[1];
  return r;
}
__device__ __forceinline__ void store_w_regs(__bf16* sW, WRegs r) {
  const int tid = threadIdx.x;
  __bf16* d = sW + (tid >> 2) * LDT + (tid & 3) * 8;
  d[0] = (__bf16)r.x.x; d[1] = (__bf16)r.x.y; d[2] = (__bf16)r.x.z; d[3] = (__bf16)r.x.w;
  d[4] = (__bf16)r.y.x; d[5] = (__bf16)r.y.y; d[6] = (__bf16)r.y.z; d[7] = (__bf16)r.y.w;
}

__device__ __forceinline__ void load_afrag(const __bf16* sA, int mwave, bf16x16 af[2]) {
#pragma unroll
  for (int i = 0; i < 2; ++i) af[i] = load_frag(sA + (mwave * 32 + i * 16) * LDT);
}

__device__ __forceinline__ void mma_tiles(const bf16x16 af[2], const __bf16* sW, int nwave,
                                          f32x8 acc[2][2]) {
  bf16x16 bfg[2];
#pragma unroll
  for (int j = 0; j < 2; ++j) bfg[j] = load_frag(sW + (nwave * 32 + j * 16) * LDT);
#pragma unroll
  for (int i = 0; i < 2; ++i)
#pragma unroll
    for (int j = 0; j < 2; ++j) acc[i][j] = wmma_bf16(af[i], bfg[j], acc[i][j]);
}

// ---------------------------------------------------------------------------
// K0: fp32 -> bf16 conversion of hidden states
__global__ __launch_bounds__(256) void cvt_bf16_kernel(const float* __restrict__ src,
                                                       __bf16* __restrict__ dst, size_t n8) {
  size_t i = (size_t)blockIdx.x * 256 + threadIdx.x;
  if (i >= n8) return;
  const float4* s = (const float4*)src + i * 2;
  float4 a = s[0], b = s[1];
  bf16x8 v;
  v[0] = (__bf16)a.x; v[1] = (__bf16)a.y; v[2] = (__bf16)a.z; v[3] = (__bf16)a.w;
  v[4] = (__bf16)b.x; v[5] = (__bf16)b.y; v[6] = (__bf16)b.z; v[7] = (__bf16)b.w;
  ((bf16x8*)dst)[i] = v;
}

// K1: router logits [BSROWS, E]; one wave per token row, lane = (e, half)
__global__ __launch_bounds__(256) void router_kernel(const __bf16* __restrict__ hs,
                                                     const float* __restrict__ gw,
                                                     float* __restrict__ logits) {
  const int wave = threadIdx.x >> 5;
  const int lane = threadIdx.x & 31;
  const size_t row = (size_t)blockIdx.x * 8 + wave;
  const int e = lane & 15;
  const int half = lane >> 4;
  const __bf16* hrow = hs + row * H_;
  const float* w = gw + (size_t)e * H_;
  float acc = 0.f;
  for (int h = half; h < H_; h += 2) acc += (float)hrow[h] * w[h];
  acc += __shfl_xor(acc, 16, 32);
  if (half == 0) logits[row * E_ + e] = acc;
}

// K2: per-(b,e) top-8 over S (softmax denom cancels -> exp-renorm over top-8)
__global__ __launch_bounds__(256) void topk_kernel(const float* __restrict__ logits,
                                                   float* __restrict__ topw,
                                                   int* __restrict__ sel) {
  __shared__ float sv[S_];
  __shared__ float rmax[256];
  __shared__ int ridx[256];
  __shared__ float selv[K_];
  __shared__ int seli[K_];
  const int be = blockIdx.x;
  const int b = be / E_, e = be % E_;
  const int tid = threadIdx.x;
  for (int s = tid; s < S_; s += 256) sv[s] = logits[((size_t)b * S_ + s) * E_ + e];
  __syncthreads();
  for (int k = 0; k < K_; ++k) {
    float best = -1e30f;
    int bi = 0;
    for (int s = tid; s < S_; s += 256)
      if (sv[s] > best) { best = sv[s]; bi = s; }
    rmax[tid] = best; ridx[tid] = bi;
    __syncthreads();
    for (int off = 128; off > 0; off >>= 1) {
      if (tid < off && rmax[tid + off] > rmax[tid]) {
        rmax[tid] = rmax[tid + off];
        ridx[tid] = ridx[tid + off];
      }
      __syncthreads();
    }
    if (tid == 0) { selv[k] = rmax[0]; seli[k] = ridx[0]; sv[ridx[0]] = -1e30f; }
    __syncthreads();
  }
  if (tid == 0) {
    float m = selv[0], sum = 0.f, w[K_];
    for (int k = 0; k < K_; ++k) { w[k] = __expf(selv[k] - m); sum += w[k]; }
    float inv = 1.f / sum;
    for (int k = 0; k < K_; ++k) {
      topw[(size_t)be * K_ + k] = w[k] * inv;
      sel[(size_t)be * K_ + k] = seli[k];
    }
  }
}

// K3: gather routed tokens into Xe [E][128][H] bf16 (row j = b*8 + k)
__global__ __launch_bounds__(256) void gather_kernel(const __bf16* __restrict__ hs,
                                                     const int* __restrict__ sel,
                                                     __bf16* __restrict__ Xe) {
  const int g = blockIdx.x;            // 0..2047
  const int e = g >> 7;
  const int j = g & 127;
  const int b = j >> 3, k = j & 7;
  const int s = sel[((size_t)b * E_ + e) * K_ + k];
  const uint2* src = (const uint2*)(hs + ((size_t)b * S_ + s) * H_);
  uint2* dst = (uint2*)(Xe + ((size_t)e * 128 + j) * H_);
  dst[threadIdx.x] = src[threadIdx.x];  // 256 * 4 bf16 = 1024
}

// K4: sgate[b,s] = sigmoid(hs . w); one wave per row
__global__ __launch_bounds__(256) void sgate_kernel(const __bf16* __restrict__ hs,
                                                    const float* __restrict__ w,
                                                    float* __restrict__ sg) {
  const int wave = threadIdx.x >> 5;
  const int lane = threadIdx.x & 31;
  const size_t row = (size_t)blockIdx.x * 8 + wave;
  const __bf16* hrow = hs + row * H_;
  float acc = 0.f;
  for (int h = lane; h < H_; h += 32) acc += (float)hrow[h] * w[h];
#pragma unroll
  for (int m = 16; m > 0; m >>= 1) acc += __shfl_xor(acc, m, 32);
  if (lane == 0) sg[row] = sigmoid_f(acc);
}

// K5: shared-expert gated GEMM + fused sgate-weighted column reduction into z[B,I]
// Double-buffered pipeline: stage k+1 (async A, reg-staged W) while WMMA on k.
__global__ __launch_bounds__(256) void gated_shared_kernel(const __bf16* __restrict__ A,
                                                           const float* __restrict__ Wg,
                                                           const float* __restrict__ Wu,
                                                           const float* __restrict__ sg,
                                                           float* __restrict__ z) {
  __shared__ __bf16 sA[2][128 * LDT];
  __shared__ __bf16 sG[2][64 * LDT];
  __shared__ __bf16 sU[2][64 * LDT];
  __shared__ float cols[64];
  const int tid = threadIdx.x;
  const int wave = tid >> 5, mwave = wave & 3, nwave = wave >> 2;
  const int rowBase = blockIdx.x * 128;
  const int colBase = blockIdx.y * 64;
  const __bf16* Ab = A + (size_t)rowBase * H_;
  const float* Wgb = Wg + (size_t)colBase * H_;
  const float* Wub = Wu + (size_t)colBase * H_;
  if (tid < 64) cols[tid] = 0.f;

  // prologue: stage k=0 into buffer 0
#if ASYNC_LDS
  stage_a_async(sA[0], Ab, H_, 0);
#else
  store_a_regs(sA[0], load_a_regs(Ab, H_, 0));
#endif
  store_w_regs(sG[0], load_w_regs(Wgb, H_, 0));
  store_w_regs(sU[0], load_w_regs(Wub, H_, 0));
  wait_async_lds();
  __syncthreads();

  f32x8 accg[2][2] = {}, accu[2][2] = {};
  int cur = 0;
  for (int k0 = 0; k0 < H_; k0 += 32) {
    const int nxt = cur ^ 1;
    const bool more = (k0 + 32) < H_;
    WRegs gn, un;
#if !ASYNC_LDS
    ARegs an;
#endif
    if (more) {
#if ASYNC_LDS
      stage_a_async(sA[nxt], Ab, H_, k0 + 32);
#else
      an = load_a_regs(Ab, H_, k0 + 32);
#endif
      gn = load_w_regs(Wgb, H_, k0 + 32);
      un = load_w_regs(Wub, H_, k0 + 32);
    }
    bf16x16 af[2];
    load_afrag(sA[cur], mwave, af);
    mma_tiles(af, sG[cur], nwave, accg);
    mma_tiles(af, sU[cur], nwave, accu);
    if (more) {
#if !ASYNC_LDS
      store_a_regs(sA[nxt], an);
#endif
      store_w_regs(sG[nxt], gn);
      store_w_regs(sU[nxt], un);
    }
    wait_async_lds();
    __syncthreads();
    cur = nxt;
  }

  const int lane = tid & 31;
  const int b = rowBase >> 11;  // 2048 rows per batch, block never straddles
#pragma unroll
  for (int j = 0; j < 2; ++j) {
    float p = 0.f;
#pragma unroll
    for (int i = 0; i < 2; ++i) {
      const int mb = mwave * 32 + i * 16 + ((lane & 16) ? 8 : 0);
#pragma unroll
      for (int r = 0; r < 8; ++r) {
        float a = silu_f(accg[i][j][r]) * accu[i][j][r];
        p += a * sg[rowBase + mb + r];
      }
    }
    atomicAdd(&cols[nwave * 32 + j * 16 + (lane & 15)], p);
  }
  __syncthreads();
  if (tid < 64) atomicAdd(&z[(size_t)b * I_ + colBase + tid], cols[tid]);
}

// K6: per-expert gated GEMM -> act [E][128][I] bf16 (same pipeline)
__global__ __launch_bounds__(256) void gated_expert_kernel(const __bf16* __restrict__ Xe,
                                                           const float* __restrict__ Wg,
                                                           const float* __restrict__ Wu,
                                                           __bf16* __restrict__ act) {
  __shared__ __bf16 sA[2][128 * LDT];
  __shared__ __bf16 sG[2][64 * LDT];
  __shared__ __bf16 sU[2][64 * LDT];
  const int tid = threadIdx.x;
  const int wave = tid >> 5, mwave = wave & 3, nwave = wave >> 2;
  const int e = blockIdx.x;
  const int colBase = blockIdx.y * 64;
  const __bf16* Ab = Xe + (size_t)e * 128 * H_;
  const float* Wgb = Wg + ((size_t)e * I_ + colBase) * H_;
  const float* Wub = Wu + ((size_t)e * I_ + colBase) * H_;

#if ASYNC_LDS
  stage_a_async(sA[0], Ab, H_, 0);
#else
  store_a_regs(sA[0], load_a_regs(Ab, H_, 0));
#endif
  store_w_regs(sG[0], load_w_regs(Wgb, H_, 0));
  store_w_regs(sU[0], load_w_regs(Wub, H_, 0));
  wait_async_lds();
  __syncthreads();

  f32x8 accg[2][2] = {}, accu[2][2] = {};
  int cur = 0;
  for (int k0 = 0; k0 < H_; k0 += 32) {
    const int nxt = cur ^ 1;
    const bool more = (k0 + 32) < H_;
    WRegs gn, un;
#if !ASYNC_LDS
    ARegs an;
#endif
    if (more) {
#if ASYNC_LDS
      stage_a_async(sA[nxt], Ab, H_, k0 + 32);
#else
      an = load_a_regs(Ab, H_, k0 + 32);
#endif
      gn = load_w_regs(Wgb, H_, k0 + 32);
      un = load_w_regs(Wub, H_, k0 + 32);
    }
    bf16x16 af[2];
    load_afrag(sA[cur], mwave, af);
    mma_tiles(af, sG[cur], nwave, accg);
    mma_tiles(af, sU[cur], nwave, accu);
    if (more) {
#if !ASYNC_LDS
      store_a_regs(sA[nxt], an);
#endif
      store_w_regs(sG[nxt], gn);
      store_w_regs(sU[nxt], un);
    }
    wait_async_lds();
    __syncthreads();
    cur = nxt;
  }

  const int lane = tid & 31;
#pragma unroll
  for (int j = 0; j < 2; ++j) {
    const int col = colBase + nwave * 32 + j * 16 + (lane & 15);
#pragma unroll
    for (int i = 0; i < 2; ++i) {
      const int mb = mwave * 32 + i * 16 + ((lane & 16) ? 8 : 0);
#pragma unroll
      for (int r = 0; r < 8; ++r) {
        float a = silu_f(accg[i][j][r]) * accu[i][j][r];
        act[((size_t)e * 128 + mb + r) * I_ + col] = (__bf16)a;
      }
    }
  }
}

// K7: per-expert down GEMM with fused top-k weighting -> expert_out [B,E,H] f32
__global__ __launch_bounds__(256) void down_expert_kernel(const __bf16* __restrict__ act,
                                                          const float* __restrict__ Wd,
                                                          const float* __restrict__ topw,
                                                          float* __restrict__ eout) {
  __shared__ __bf16 sA[2][128 * LDT];
  __shared__ __bf16 sW[2][64 * LDT];
  const int tid = threadIdx.x;
  const int wave = tid >> 5, mwave = wave & 3, nwave = wave >> 2;
  const int e = blockIdx.x;
  const int colBase = blockIdx.y * 64;
  const __bf16* Ab = act + (size_t)e * 128 * I_;
  const float* Wb = Wd + ((size_t)e * H_ + colBase) * I_;

#if ASYNC_LDS
  stage_a_async(sA[0], Ab, I_, 0);
#else
  store_a_regs(sA[0], load_a_regs(Ab, I_, 0));
#endif
  store_w_regs(sW[0], load_w_regs(Wb, I_, 0));
  wait_async_lds();
  __syncthreads();

  f32x8 acc[2][2] = {};
  int cur = 0;
  for (int k0 = 0; k0 < I_; k0 += 32) {
    const int nxt = cur ^ 1;
    const bool more = (k0 + 32) < I_;
    WRegs wn;
#if !ASYNC_LDS
    ARegs an;
#endif
    if (more) {
#if ASYNC_LDS
      stage_a_async(sA[nxt], Ab, I_, k0 + 32);
#else
      an = load_a_regs(Ab, I_, k0 + 32);
#endif
      wn = load_w_regs(Wb, I_, k0 + 32);
    }
    bf16x16 af[2];
    load_afrag(sA[cur], mwave, af);
    mma_tiles(af, sW[cur], nwave, acc);
    if (more) {
#if !ASYNC_LDS
      store_a_regs(sA[nxt], an);
#endif
      store_w_regs(sW[nxt], wn);
    }
    wait_async_lds();
    __syncthreads();
    cur = nxt;
  }

  const int lane = tid & 31;
#pragma unroll
  for (int j = 0; j < 2; ++j) {
    const int h = colBase + nwave * 32 + j * 16 + (lane & 15);
#pragma unroll
    for (int i = 0; i < 2; ++i) {
      const int mb = mwave * 32 + i * 16 + ((lane & 16) ? 8 : 0);
      const int b = mb >> 3;  // rows mb..mb+7 are exactly (b, k=0..7)
      const float* tw = topw + ((size_t)b * E_ + e) * K_;
      float s = 0.f;
#pragma unroll
      for (int r = 0; r < 8; ++r) s += tw[r] * acc[i][j][r];
      eout[((size_t)b * E_ + e) * H_ + h] = s;
    }
  }
}

// K8: shared[b,h] = sum_i z[b,i] * Wd[h,i]  (tiny GEMV, 92 MFLOP)
__global__ __launch_bounds__(256) void shared_down_kernel(const float* __restrict__ z,
                                                          const float* __restrict__ Wd,
                                                          float* __restrict__ sh) {
  const int idx = blockIdx.x * 256 + threadIdx.x;  // B_*H_ = 16384
  const int b = idx >> 10, h = idx & 1023;
  const float* zr = z + (size_t)b * I_;
  const float* wr = Wd + (size_t)h * I_;
  float acc = 0.f;
  for (int i = 0; i < I_; ++i) acc += zr[i] * wr[i];
  sh[(size_t)b * H_ + h] = acc;
}

// K9: build concat matrix F [B*E, 3H] bf16 = [expert_out | shared | cls]
__global__ __launch_bounds__(256) void build_f_kernel(const float* __restrict__ eout,
                                                      const float* __restrict__ sh,
                                                      const __bf16* __restrict__ hs,
                                                      __bf16* __restrict__ F) {
  const int row = blockIdx.x;       // b*E + e
  const int b = row >> 4;
  __bf16* dst = F + (size_t)row * (3 * H_);
  for (int h = threadIdx.x; h < H_; h += 256) {
    dst[h] = (__bf16)eout[(size_t)row * H_ + h];
    dst[H_ + h] = (__bf16)sh[(size_t)b * H_ + h];
    dst[2 * H_ + h] = hs[(size_t)b * S_ * H_ + h];  // s = 0 (cls token)
  }
}

// K10: generic C = A(bf16,[M,K]) x W(f32,[N,K])^T + bias; optional bf16 mirror
__global__ __launch_bounds__(256) void gemm_nt_bias_kernel(const __bf16* __restrict__ A,
                                                           const float* __restrict__ W,
                                                           const float* __restrict__ bias,
                                                           float* __restrict__ C,
                                                           __bf16* __restrict__ Cb,
                                                           int Kd, int N) {
  __shared__ __bf16 sA[2][128 * LDT];
  __shared__ __bf16 sW[2][64 * LDT];
  const int tid = threadIdx.x;
  const int wave = tid >> 5, mwave = wave & 3, nwave = wave >> 2;
  const int rowBase = blockIdx.x * 128;
  const int colBase = blockIdx.y * 64;
  const __bf16* Ab = A + (size_t)rowBase * Kd;
  const float* Wb = W + (size_t)colBase * Kd;

#if ASYNC_LDS
  stage_a_async(sA[0], Ab, Kd, 0);
#else
  store_a_regs(sA[0], load_a_regs(Ab, Kd, 0));
#endif
  store_w_regs(sW[0], load_w_regs(Wb, Kd, 0));
  wait_async_lds();
  __syncthreads();

  f32x8 acc[2][2] = {};
  int cur = 0;
  for (int k0 = 0; k0 < Kd; k0 += 32) {
    const int nxt = cur ^ 1;
    const bool more = (k0 + 32) < Kd;
    WRegs wn;
#if !ASYNC_LDS
    ARegs an;
#endif
    if (more) {
#if ASYNC_LDS
      stage_a_async(sA[nxt], Ab, Kd, k0 + 32);
#else
      an = load_a_regs(Ab, Kd, k0 + 32);
#endif
      wn = load_w_regs(Wb, Kd, k0 + 32);
    }
    bf16x16 af[2];
    load_afrag(sA[cur], mwave, af);
    mma_tiles(af, sW[cur], nwave, acc);
    if (more) {
#if !ASYNC_LDS
      store_a_regs(sA[nxt], an);
#endif
      store_w_regs(sW[nxt], wn);
    }
    wait_async_lds();
    __syncthreads();
    cur = nxt;
  }

  const int lane = tid & 31;
#pragma unroll
  for (int j = 0; j < 2; ++j) {
    const int col = colBase + nwave * 32 + j * 16 + (lane & 15);
    const float bv = bias[col];
#pragma unroll
    for (int i = 0; i < 2; ++i) {
      const int mb = mwave * 32 + i * 16 + ((lane & 16) ? 8 : 0);
#pragma unroll
      for (int r = 0; r < 8; ++r) {
        const int row = rowBase + mb + r;
        float c = acc[i][j][r] + bv;
        C[(size_t)row * N + col] = c;
        if (Cb) Cb[(size_t)row * N + col] = (__bf16)c;
      }
    }
  }
}

// ---------------------------------------------------------------------------
extern "C" void kernel_launch(void* const* d_in, const int* in_sizes, int n_in,
                              void* d_out, int out_size, void* d_ws, size_t ws_size,
                              hipStream_t stream) {
  (void)in_sizes; (void)n_in; (void)out_size; (void)ws_size;
  const float* hs          = (const float*)d_in[0];
  const float* gate_w      = (const float*)d_in[1];
  const float* exp_gate_w  = (const float*)d_in[2];
  const float* exp_up_w    = (const float*)d_in[3];
  const float* exp_down_w  = (const float*)d_in[4];
  const float* sh_gate_w   = (const float*)d_in[5];
  const float* sh_up_w     = (const float*)d_in[6];
  const float* sh_down_w   = (const float*)d_in[7];
  const float* sh_exp_gate = (const float*)d_in[8];
  const float* feature_w   = (const float*)d_in[9];
  const float* feature_b   = (const float*)d_in[10];
  const float* out_w       = (const float*)d_in[11];
  const float* out_b       = (const float*)d_in[12];
  float* out = (float*)d_out;

  char* ws = (char*)d_ws;
  size_t off = 0;
  auto take = [&](size_t bytes) -> char* {
    char* p = ws + off;
    off = (off + bytes + 255) & ~(size_t)255;
    return p;
  };
  __bf16* hs_bf  = (__bf16*)take((size_t)BSROWS * H_ * sizeof(__bf16));
  float*  logits = (float*)take((size_t)BSROWS * E_ * sizeof(float));
  float*  topw   = (float*)take((size_t)B_ * E_ * K_ * sizeof(float));
  int*    sel    = (int*)take((size_t)B_ * E_ * K_ * sizeof(int));
  __bf16* Xe     = (__bf16*)take((size_t)E_ * 128 * H_ * sizeof(__bf16));
  float*  sg     = (float*)take((size_t)BSROWS * sizeof(float));
  float*  zacc   = (float*)take((size_t)B_ * I_ * sizeof(float));
  __bf16* act    = (__bf16*)take((size_t)E_ * 128 * I_ * sizeof(__bf16));
  float*  eout   = (float*)take((size_t)B_ * E_ * H_ * sizeof(float));
  float*  shout  = (float*)take((size_t)B_ * H_ * sizeof(float));
  __bf16* Fm     = (__bf16*)take((size_t)B_ * E_ * 3 * H_ * sizeof(__bf16));
  float*  featf  = (float*)take((size_t)B_ * E_ * H_ * sizeof(float));
  __bf16* featb  = (__bf16*)take((size_t)B_ * E_ * H_ * sizeof(__bf16));

  cvt_bf16_kernel<<<(BSROWS * H_ / 8 + 255) / 256, 256, 0, stream>>>(
      hs, hs_bf, (size_t)BSROWS * H_ / 8);
  router_kernel<<<BSROWS / 8, 256, 0, stream>>>(hs_bf, gate_w, logits);
  topk_kernel<<<B_ * E_, 256, 0, stream>>>(logits, topw, sel);
  gather_kernel<<<E_ * B_ * K_, 256, 0, stream>>>(hs_bf, sel, Xe);
  sgate_kernel<<<BSROWS / 8, 256, 0, stream>>>(hs_bf, sh_exp_gate, sg);
  (void)hipMemsetAsync(zacc, 0, (size_t)B_ * I_ * sizeof(float), stream);
  gated_shared_kernel<<<dim3(BSROWS / 128, I_ / 64), 256, 0, stream>>>(
      hs_bf, sh_gate_w, sh_up_w, sg, zacc);
  gated_expert_kernel<<<dim3(E_, I_ / 64), 256, 0, stream>>>(
      Xe, exp_gate_w, exp_up_w, act);
  down_expert_kernel<<<dim3(E_, H_ / 64), 256, 0, stream>>>(
      act, exp_down_w, topw, eout);
  shared_down_kernel<<<(B_ * H_) / 256, 256, 0, stream>>>(zacc, sh_down_w, shout);
  build_f_kernel<<<B_ * E_, 256, 0, stream>>>(eout, shout, hs_bf, Fm);
  gemm_nt_bias_kernel<<<dim3((B_ * E_) / 128, H_ / 64), 256, 0, stream>>>(
      Fm, feature_w, feature_b, featf, featb, 3 * H_, H_);
  gemm_nt_bias_kernel<<<dim3((B_ * E_) / 128, T_ / 64), 256, 0, stream>>>(
      featb, out_w, out_b, out, (__bf16*)nullptr, H_, T_);
}
